// AMMN_Net_49624052138586
// MI455X (gfx1250) — matrix operand order repaired and checked by reading, not compiled, not run on hardware
//
#include <hip/hip_runtime.h>
#include <hip/hip_bf16.h>
#include <math.h>

#define NN 100000
#define DIN 128
#define DH  64
#define DE  64

typedef __attribute__((ext_vector_type(16))) _Float16 v16h;
typedef __attribute__((ext_vector_type(8)))  float    v8f;

__device__ __forceinline__ float fast_tanh(float v) {
  float e = __expf(2.0f * v);           // v_exp_f32; e->inf => 1, e->0 => -1
  return 1.0f - 2.0f / (e + 1.0f);
}

// ---- pack W[K,N] (f32, row-major) into WMMA B-fragment layout (f16) ----
// PB[(kc*(N/16) + t)*32 + lane][16]: lane<16 -> col=t*16+lane, K=k0..k0+7,k0+16..k0+23;
// lane>=16 -> same col, K=k0+8..k0+15,k0+24..k0+31  (ISA 7.12.2 B layout for 16x16x32).
__global__ void pack_b_kernel(const float* __restrict__ W, _Float16* __restrict__ PB,
                              int K, int N) {
  int idx = blockIdx.x * blockDim.x + threadIdx.x;
  int total = (K >> 5) * (N >> 4) * 32;
  if (idx >= total) return;
  int lane = idx & 31;
  int tt   = (idx >> 5) % (N >> 4);
  int kc   = (idx >> 5) / (N >> 4);
  int col  = tt * 16 + (lane & 15);
  int k0   = kc * 32 + ((lane >> 4) << 3);
  _Float16* out = PB + (size_t)idx * 16;
#pragma unroll
  for (int t = 0; t < 16; ++t)
    out[t] = (_Float16)W[(size_t)(k0 + t + ((t >> 3) << 3)) * N + col];
}

__device__ __forceinline__ v16h load_b_packed(const _Float16* __restrict__ PB,
                                              int k0, int N, int t, int lane) {
  return *(const v16h*)(PB + ((((size_t)(k0 >> 5) * (N >> 4) + t) << 5) << 4) + (lane << 4));
}

// A-fragment (16x32 f16) from f32 row; optional fused relu(batchnorm) via scale/shift.
template <bool BN>
__device__ __forceinline__ v16h load_a_frag(const float* __restrict__ arow, int k0, int kbA,
                                            const float* __restrict__ scale,
                                            const float* __restrict__ shift) {
  float4 p0 = *(const float4*)(arow + k0 + kbA);
  float4 p1 = *(const float4*)(arow + k0 + kbA + 4);
  float4 p2 = *(const float4*)(arow + k0 + 16 + kbA);
  float4 p3 = *(const float4*)(arow + k0 + 16 + kbA + 4);
  float va[16] = {p0.x, p0.y, p0.z, p0.w, p1.x, p1.y, p1.z, p1.w,
                  p2.x, p2.y, p2.z, p2.w, p3.x, p3.y, p3.z, p3.w};
  v16h a;
#pragma unroll
  for (int i = 0; i < 16; ++i) {
    float v = va[i];
    if (BN) {
      int kidx = (i < 8) ? (k0 + kbA + i) : (k0 + 16 + kbA + (i - 8));
      v = fmaxf(v * scale[kidx] + shift[kidx], 0.0f);
    }
    a[i] = (_Float16)v;
  }
  return a;
}

// ---- layer-1 GEMM for BOTH branches: each 16-row slab of x read exactly once. ----
__global__ __launch_bounds__(32)
void gemm1_dual_kernel(const float* __restrict__ A,
                       const _Float16* __restrict__ PBg, const _Float16* __restrict__ PBs,
                       float* __restrict__ Cg, float* __restrict__ Cs) {
  const int lane = threadIdx.x;
  const int m0 = blockIdx.x * 16;
  const int cl = lane & 15;
  const int kbA = (lane >> 4) << 3;
  const float* arow = A + (size_t)(m0 + cl) * DIN;
  v8f accg[4], accs[4];
#pragma unroll
  for (int t = 0; t < 4; ++t) { accg[t] = 0.0f; accs[t] = 0.0f; }
  for (int k0 = 0; k0 < DIN; k0 += 32) {
    __builtin_prefetch(arow + k0 + 32, 0, 1);      // global_prefetch_b8 (speculative)
    v16h a = load_a_frag<false>(arow, k0, kbA, nullptr, nullptr);
#pragma unroll
    for (int t = 0; t < 4; ++t) {
      v16h b = load_b_packed(PBg, k0, DH, t, lane);
      accg[t] = __builtin_amdgcn_wmma_f32_16x16x32_f16(false, a, false, b,
                                                       (short)0, accg[t], false, false);
    }
#pragma unroll
    for (int t = 0; t < 4; ++t) {
      v16h b = load_b_packed(PBs, k0, DH, t, lane);
      accs[t] = __builtin_amdgcn_wmma_f32_16x16x32_f16(false, a, false, b,
                                                       (short)0, accs[t], false, false);
    }
  }
  const int rb = (lane >> 4) << 3;
#pragma unroll
  for (int t = 0; t < 4; ++t) {
    int col = t * 16 + cl;
#pragma unroll
    for (int r = 0; r < 8; ++r) {
      Cg[(size_t)(m0 + rb + r) * DH + col] = accg[t][r];
      Cs[(size_t)(m0 + rb + r) * DH + col] = accs[t][r];
    }
  }
}

// ---- generic WMMA GEMM: one wave = 16 rows x (NT*16) cols (full N extent). ----
template <int NT, bool BN, bool TANH>
__global__ __launch_bounds__(32)
void gemm_wmma_kernel(const float* __restrict__ A, const _Float16* __restrict__ PB,
                      float* __restrict__ C, int K, int N,
                      const float* __restrict__ scale, const float* __restrict__ shift) {
  const int lane = threadIdx.x;
  const int m0 = blockIdx.x * 16;
  const int cl = lane & 15;
  const int kbA = (lane >> 4) << 3;
  const float* arow = A + (size_t)(m0 + cl) * K;
  v8f acc[NT];
#pragma unroll
  for (int t = 0; t < NT; ++t) acc[t] = 0.0f;
  for (int k0 = 0; k0 < K; k0 += 32) {
    __builtin_prefetch(arow + k0 + 32, 0, 1);
    v16h a = load_a_frag<BN>(arow, k0, kbA, scale, shift);
#pragma unroll
    for (int t = 0; t < NT; ++t) {
      v16h b = load_b_packed(PB, k0, N, t, lane);
      acc[t] = __builtin_amdgcn_wmma_f32_16x16x32_f16(false, a, false, b,
                                                      (short)0, acc[t], false, false);
    }
  }
  const int rb = (lane >> 4) << 3;
#pragma unroll
  for (int t = 0; t < NT; ++t) {
    int col = t * 16 + cl;
#pragma unroll
    for (int r = 0; r < 8; ++r) {
      float v = acc[t][r];
      if (TANH) v = fast_tanh(v);
      C[(size_t)(m0 + rb + r) * N + col] = v;
    }
  }
}

// ---------------- degree / dinv ----------------
__global__ void deg_init_kernel(float* deg, int n) {
  int i = blockIdx.x * blockDim.x + threadIdx.x;
  if (i < n) deg[i] = 1.0f;                      // self-loop
}
__global__ void deg_edges_kernel(const int* __restrict__ dst, float* deg, int e) {
  int i = blockIdx.x * blockDim.x + threadIdx.x;
  if (i < e) atomicAdd(&deg[dst[i]], 1.0f);
}
__global__ void deg_fin_kernel(float* dinv, int n) {
  int i = blockIdx.x * blockDim.x + threadIdx.x;
  if (i < n) dinv[i] = rsqrtf(dinv[i]);          // deg >= 1 always
}

// ---- aggregation: out = segsum_{e:dst=d}(h[src]*dinv[src]*dinv[d]) + h[d]*dinv[d]^2 + b ----
__global__ void agg_init_kernel(const float* __restrict__ h, const float* __restrict__ dinv,
                                const float* __restrict__ bias, float* __restrict__ out, int n) {
  int idx = blockIdx.x * blockDim.x + threadIdx.x;
  if (idx < n * DH) {
    int i = idx >> 6, c = idx & 63;
    float d = dinv[i];
    out[idx] = h[idx] * d * d + bias[c];
  }
}
__global__ void agg_edges_kernel(const float* __restrict__ h, const int* __restrict__ src,
                                 const int* __restrict__ dst, const float* __restrict__ dinv,
                                 float* __restrict__ out, int e) {
  long long tid = (long long)blockIdx.x * blockDim.x + threadIdx.x;
  int ei = (int)(tid >> 4);                      // 16 threads per edge, 4 cols each
  int q  = (int)(tid & 15);
  if (ei < e) {
    int s = src[ei], d = dst[ei];
    float w = dinv[s] * dinv[d];
    const float4 hv = *(const float4*)(h + (size_t)s * DH + q * 4);
    float* o = out + (size_t)d * DH + q * 4;
    atomicAdd(o + 0, hv.x * w);
    atomicAdd(o + 1, hv.y * w);
    atomicAdd(o + 2, hv.z * w);
    atomicAdd(o + 3, hv.w * w);
  }
}

// ---- BatchNorm training stats -> fused scale/shift: y = max(v*scale + shift, 0) ----
__global__ __launch_bounds__(256)
void bn_stats_kernel(const float* __restrict__ h, const float* __restrict__ gamma,
                     const float* __restrict__ beta, float* __restrict__ scale,
                     float* __restrict__ shift, int n) {
  __shared__ float ssum[256], ssq[256];
  int c = blockIdx.x;                            // one block per column
  float s = 0.f, s2 = 0.f;
  for (int i = threadIdx.x; i < n; i += 256) {
    float v = h[(size_t)i * DH + c];
    s += v; s2 += v * v;
  }
  ssum[threadIdx.x] = s; ssq[threadIdx.x] = s2;
  __syncthreads();
  for (int off = 128; off > 0; off >>= 1) {
    if (threadIdx.x < off) {
      ssum[threadIdx.x] += ssum[threadIdx.x + off];
      ssq[threadIdx.x]  += ssq[threadIdx.x + off];
    }
    __syncthreads();
  }
  if (threadIdx.x == 0) {
    float m = ssum[0] / n;
    float var = ssq[0] / n - m * m;
    float rsg = rsqrtf(var + 1e-5f) * gamma[c];
    scale[c] = rsg;
    shift[c] = beta[c] - m * rsg;
  }
}

// ---------------- head: gather concat [node_u | node_i] ----------------
__global__ void gather_cat_kernel(const float* __restrict__ nodes, const int* __restrict__ users,
                                  const int* __restrict__ items, float* __restrict__ out, int b) {
  int idx = blockIdx.x * blockDim.x + threadIdx.x;
  if (idx < b * 2 * DE) {
    int r = idx >> 7, j = idx & 127;
    int node = (j < DE) ? users[r] : items[r];
    out[idx] = nodes[(size_t)node * DE + (j & 63)];
  }
}

// ---------------- head: gate + fusion + output heads ----------------
__global__ void head_final_kernel(const float* __restrict__ gen, const float* __restrict__ h1,
                                  const float* __restrict__ h2, const float* __restrict__ Wsig,
                                  const float* __restrict__ Wtc, const float* __restrict__ btc,
                                  const float* __restrict__ Wp, const float* __restrict__ bp,
                                  float* __restrict__ link_out, float* __restrict__ gen_out, int b) {
  int r = blockIdx.x * blockDim.x + threadIdx.x;
  if (r >= b) return;
  const float* h1r = h1 + (size_t)r * 128;
  const float* h2r = h2 + (size_t)r * 128;
  float za = 0.f;
  for (int j = 0; j < 128; ++j) za += h1r[j] * Wsig[j];
  for (int j = 0; j < 128; ++j) za += h2r[j] * Wsig[128 + j];
  float z = 1.0f / (1.0f + __expf(-za));
  float l0 = bp[0], l1 = bp[1];
  for (int j = 0; j < 128; ++j) {
    float f = z * h1r[j] + (1.f - z) * h2r[j];
    l0 += f * Wp[j * 2 + 0];
    l1 += f * Wp[j * 2 + 1];
  }
  link_out[r * 2 + 0] = l0;
  link_out[r * 2 + 1] = l1;
  const float* gr = gen + (size_t)r * 128;
  float t0 = btc[0], t1 = btc[1], t2 = btc[2], t3 = btc[3];
  for (int j = 0; j < 128; ++j) {
    float g = gr[j];
    t0 += g * Wtc[j * 4 + 0]; t1 += g * Wtc[j * 4 + 1];
    t2 += g * Wtc[j * 4 + 2]; t3 += g * Wtc[j * 4 + 3];
  }
  gen_out[r * 4 + 0] = t0; gen_out[r * 4 + 1] = t1;
  gen_out[r * 4 + 2] = t2; gen_out[r * 4 + 3] = t3;
}

extern "C" void kernel_launch(void* const* d_in, const int* in_sizes, int n_in,
                              void* d_out, int out_size, void* d_ws, size_t ws_size,
                              hipStream_t stream) {
  const float* x     = (const float*)d_in[0];
  const int*   src   = (const int*)d_in[1];
  const int    E     = in_sizes[1] / 2;
  const int*   dst   = src + E;
  const int*   users = (const int*)d_in[2];
  const int*   items = (const int*)d_in[3];
  const int    B     = in_sizes[2];
  const float* Wf1   = (const float*)d_in[16];
  const float* Wsig  = (const float*)d_in[17];
  const float* Wtc   = (const float*)d_in[18];
  const float* btc   = (const float*)d_in[19];
  const float* Wp    = (const float*)d_in[20];
  const float* bp    = (const float*)d_in[21];

  float* ws = (float*)d_ws;
  const size_t NF     = (size_t)NN * DH;            // 6.4M floats per node matrix
  const size_t oDinv  = 0;
  const size_t oBufA  = 100352;                     // branch-g h (layer outputs)
  const size_t oBufB  = oBufA + NF;                 // branch-s h
  const size_t oBufC  = oBufB + NF;                 // aggregation scratch
  const size_t oGen   = oBufC + NF;
  const size_t oSpec  = oGen + NF;
  const size_t oScale = oSpec + NF;
  const size_t oShift = oScale + 64;
  const size_t oPack  = oShift + 64;                // packed f16 weights (5 x 16384 halves)
  float* dinv  = ws + oDinv;
  float* bufA  = ws + oBufA;
  float* bufB  = ws + oBufB;
  float* bufC  = ws + oBufC;
  float* scale = ws + oScale;
  float* shift = ws + oShift;
  _Float16* pb = (_Float16*)(ws + oPack);
  _Float16* PB1g = pb;                              // W1 pack: 4*4*32*16 = 8192 halves
  _Float16* PB1s = pb + 16384;
  _Float16* PB2g = pb + 32768;                      // W2 pack: 2*4*32*16 = 4096 halves
  _Float16* PB2s = pb + 49152;
  _Float16* PBf  = pb + 65536;                      // Wf1 pack: 4*8*32*16 = 16384 halves
  // overlays for the head stage (bufA/bufB free by then)
  float* gen_cat  = ws + oBufA;
  float* spec_cat = ws + oBufA + (size_t)B * 128;
  float* h1t      = ws + oBufB;
  float* h2t      = ws + oBufB + (size_t)B * 128;

  const int T = 256;
  // --- pack all weight matrices into WMMA B-fragment layout (tiny, L2-resident) ---
  pack_b_kernel<<<2, T, 0, stream>>>((const float*)d_in[4],  PB1g, DIN, DH);
  pack_b_kernel<<<2, T, 0, stream>>>((const float*)d_in[10], PB1s, DIN, DH);
  pack_b_kernel<<<1, T, 0, stream>>>((const float*)d_in[8],  PB2g, DH,  DE);
  pack_b_kernel<<<1, T, 0, stream>>>((const float*)d_in[14], PB2s, DH,  DE);
  pack_b_kernel<<<4, T, 0, stream>>>(Wf1, PBf, 128, 128);

  // --- symmetric normalization coefficients ---
  deg_init_kernel<<<(NN + T - 1) / T, T, 0, stream>>>(dinv, NN);
  deg_edges_kernel<<<(E + T - 1) / T, T, 0, stream>>>(dst, dinv, E);
  deg_fin_kernel<<<(NN + T - 1) / T, T, 0, stream>>>(dinv, NN);

  // --- layer 1 for both branches, x read once ---
  gemm1_dual_kernel<<<NN / 16, 32, 0, stream>>>(x, PB1g, PB1s, bufA, bufB);

  // --- per branch: aggregate -> BN stats -> (BN+ReLU fused) GEMM2 -> aggregate ---
  for (int br = 0; br < 2; ++br) {
    const int base = 4 + br * 6;
    const float* b1  = (const float*)d_in[base + 1];
    const float* g1  = (const float*)d_in[base + 2];
    const float* be1 = (const float*)d_in[base + 3];
    const float* b2  = (const float*)d_in[base + 5];
    _Float16* PB2 = (br == 0) ? PB2g : PB2s;
    float* h1buf = (br == 0) ? bufA : bufB;           // layer-1 output (this branch)
    float* nodes = ws + (br == 0 ? oGen : oSpec);

    agg_init_kernel<<<(NN * DH + T - 1) / T, T, 0, stream>>>(h1buf, dinv, b1, bufC, NN);
    agg_edges_kernel<<<(((long long)E * 16 + T - 1) / T), T, 0, stream>>>(h1buf, src, dst,
                                                                          dinv, bufC, E);
    bn_stats_kernel<<<DH, 256, 0, stream>>>(bufC, g1, be1, scale, shift, NN);
    gemm_wmma_kernel<4, true, false><<<NN / 16, 32, 0, stream>>>(bufC, PB2, h1buf, DH, DE,
                                                                 scale, shift);
    agg_init_kernel<<<(NN * DE + T - 1) / T, T, 0, stream>>>(h1buf, dinv, b2, nodes, NN);
    agg_edges_kernel<<<(((long long)E * 16 + T - 1) / T), T, 0, stream>>>(h1buf, src, dst,
                                                                          dinv, nodes, E);
  }

  // --- fusion head ---
  gather_cat_kernel<<<(B * 128 + T - 1) / T, T, 0, stream>>>(ws + oGen,  users, items, gen_cat,  B);
  gather_cat_kernel<<<(B * 128 + T - 1) / T, T, 0, stream>>>(ws + oSpec, users, items, spec_cat, B);
  gemm_wmma_kernel<8, false, true><<<B / 16, 32, 0, stream>>>(gen_cat,  PBf, h1t, 128, 128,
                                                              nullptr, nullptr);
  gemm_wmma_kernel<8, false, true><<<B / 16, 32, 0, stream>>>(spec_cat, PBf, h2t, 128, 128,
                                                              nullptr, nullptr);

  float* link_out = (float*)d_out;                  // [B,2]
  float* gen_out  = (float*)d_out + (size_t)B * 2;  // [B,4]
  head_final_kernel<<<(B + T - 1) / T, T, 0, stream>>>(gen_cat, h1t, h2t, Wsig, Wtc, btc,
                                                       Wp, bp, link_out, gen_out, B);
}
